// PointTransformerV2_x_63479616634994
// MI455X (gfx1250) — compile-verified, matrix operand order checked by reading, not run on hardware
//
#include <hip/hip_runtime.h>

#define NPTS 32768
#define CH   256
#define NGRP 8
#define NSMP 16

typedef __attribute__((ext_vector_type(16))) __bf16 v16bf;
typedef __attribute__((ext_vector_type(8)))  __bf16 v8bf;
typedef __attribute__((ext_vector_type(8)))  float  v8f;

union AFrag { v16bf v; v8bf h[2]; };

__device__ __forceinline__ v8f wmma_bf16(const AFrag& a, const AFrag& b, v8f c) {
  return __builtin_amdgcn_wmma_f32_16x16x32_bf16(false, a.v, false, b.v, (short)0, c, false, false);
}

// ---------------------------------------------------------------------------
// Convert the 4 big f32 weights to bf16, column-major (B-operand layout:
// dst[n*CH + k] = W[k][n]).  blockIdx.y selects {Wq, Wk, Wv, Wp2}.
// ---------------------------------------------------------------------------
__global__ void cvt_w_kernel(const float* __restrict__ Wq, const float* __restrict__ Wk,
                             const float* __restrict__ Wv, const float* __restrict__ Wp2,
                             __bf16* __restrict__ dst) {
  const float* srcs[4] = {Wq, Wk, Wv, Wp2};
  const float* s = srcs[blockIdx.y];
  __bf16* d = dst + (size_t)blockIdx.y * CH * CH;
  int n = blockIdx.x;
  int k = threadIdx.x;
  d[(size_t)n * CH + k] = (__bf16)s[(size_t)k * CH + n];
}

// ---------------------------------------------------------------------------
// Projection: out = [ReLU(LN(x@W + b))] for q,k ; x@W + b for v.
// One block = 16 rows. 8 waves, each wave computes two 16x16 col tiles via
// v_wmma_f32_16x16x32_bf16, K=256 in 8 steps. LN fused in LDS.
// blockIdx.y: 0=q, 1=k, 2=v.
// ---------------------------------------------------------------------------
__global__ void __launch_bounds__(256)
proj_kernel(const float* __restrict__ q, const float* __restrict__ k,
            const float* __restrict__ v, const __bf16* __restrict__ Wbf,
            const float* __restrict__ bq, const float* __restrict__ bk,
            const float* __restrict__ bv,
            const float* __restrict__ gq, const float* __restrict__ betaq,
            const float* __restrict__ gk, const float* __restrict__ betak,
            float* __restrict__ outq, float* __restrict__ outk,
            float* __restrict__ outv) {
  __shared__ __align__(16) __bf16 s_a[16][264];
  __shared__ float s_o[16][256];
  __shared__ float s_red[2][16][16];
  __shared__ float s_mu[16], s_rs[16];

  const int which = blockIdx.y;
  const float*  x     = (which == 0) ? q : ((which == 1) ? k : v);
  const __bf16* W     = Wbf + (size_t)which * CH * CH;
  const float*  bias  = (which == 0) ? bq : ((which == 1) ? bk : bv);
  const float*  gamma = (which == 0) ? gq : gk;
  const float*  beta  = (which == 0) ? betaq : betak;
  float*        out   = (which == 0) ? outq : ((which == 1) ? outk : outv);
  const bool do_ln = (which < 2);

  const int tid = threadIdx.x;
  const int n0  = blockIdx.x * 16;
  const int r   = tid >> 4, seg = tid & 15;

  // load 16 input rows, convert f32 -> bf16 into LDS
  #pragma unroll
  for (int i = 0; i < 16; ++i) {
    int c = seg * 16 + i;
    s_a[r][c] = (__bf16)x[(size_t)(n0 + r) * CH + c];
  }
  __syncthreads();

  const int lane = tid & 31, wave = tid >> 5;
  const int m    = lane & 15;
  const int koff = (lane & 16) ? 8 : 0;
  const int nc0  = wave * 32 + m;
  const int nc1  = nc0 + 16;
  v8f acc0 = {}, acc1 = {};
  #pragma unroll
  for (int ks = 0; ks < 8; ++ks) {
    int kb = ks * 32 + koff;
    AFrag a, b0, b1;
    a.h[0]  = *(const v8bf*)&s_a[m][kb];
    a.h[1]  = *(const v8bf*)&s_a[m][kb + 16];
    b0.h[0] = *(const v8bf*)&W[(size_t)nc0 * CH + kb];
    b0.h[1] = *(const v8bf*)&W[(size_t)nc0 * CH + kb + 16];
    b1.h[0] = *(const v8bf*)&W[(size_t)nc1 * CH + kb];
    b1.h[1] = *(const v8bf*)&W[(size_t)nc1 * CH + kb + 16];
    acc0 = wmma_bf16(a, b0, acc0);
    acc1 = wmma_bf16(a, b1, acc1);
  }
  const int orow = (lane & 16) ? 8 : 0;
  #pragma unroll
  for (int rr = 0; rr < 8; ++rr) {
    s_o[orow + rr][nc0] = acc0[rr] + bias[nc0];
    s_o[orow + rr][nc1] = acc1[rr] + bias[nc1];
  }
  __syncthreads();

  if (do_ln) {
    float sum = 0.f, sq = 0.f;
    #pragma unroll
    for (int i = 0; i < 16; ++i) {
      float t = s_o[r][seg * 16 + i];
      sum += t; sq += t * t;
    }
    s_red[0][r][seg] = sum;
    s_red[1][r][seg] = sq;
    __syncthreads();
    if (tid < 16) {
      float s1 = 0.f, s2 = 0.f;
      #pragma unroll
      for (int j = 0; j < 16; ++j) { s1 += s_red[0][tid][j]; s2 += s_red[1][tid][j]; }
      float mu = s1 * (1.f / CH);
      float var = s2 * (1.f / CH) - mu * mu;
      s_mu[tid] = mu;
      s_rs[tid] = rsqrtf(var + 1e-5f);
    }
    __syncthreads();
    float mu = s_mu[r], rs = s_rs[r];
    #pragma unroll
    for (int i = 0; i < 16; ++i) {
      int c = seg + 16 * i;
      float y = (s_o[r][c] - mu) * rs * gamma[c] + beta[c];
      out[(size_t)(n0 + r) * CH + c] = fmaxf(y, 0.f);
    }
  } else {
    #pragma unroll
    for (int i = 0; i < 16; ++i) {
      int c = seg + 16 * i;
      out[(size_t)(n0 + r) * CH + c] = s_o[r][c];
    }
  }
}

// ---------------------------------------------------------------------------
// Fused grouped-vector-attention: one block per point (NS=16 neighbor rows).
// Heavy part: peb = ReLU(LN(pos@Wp1+b)) @ Wp2 + b  (16x256x256 per point) via
// WMMA bf16; everything else (gather, relation, 256->8 weight encoding, LN,
// softmax, grouped einsum) stays in LDS.
// ---------------------------------------------------------------------------
__global__ void __launch_bounds__(256)
attn_kernel(const float* __restrict__ query, const float* __restrict__ keyf,
            const float* __restrict__ val,   const float* __restrict__ xyz,
            const int* __restrict__ refidx,
            const float* __restrict__ Wp1, const float* __restrict__ bp1,
            const float* __restrict__ gp,  const float* __restrict__ betap,
            const __bf16* __restrict__ Wp2bf, const float* __restrict__ bp2,
            const float* __restrict__ Ww1, const float* __restrict__ bw1,
            const float* __restrict__ gw,  const float* __restrict__ betaw,
            const float* __restrict__ Ww2, const float* __restrict__ bw2,
            float* __restrict__ out) {
  __shared__ __align__(16) __bf16 s_hid[16][264];
  __shared__ float s_peb[16][256];   // later reused as relation
  __shared__ float s_valg[16][256];
  __shared__ float s_red[2][16][16];
  __shared__ float s_mu[16], s_rs[16];
  __shared__ float s_pos[16][3];
  __shared__ float s_mask[16];
  __shared__ int   s_idx[16];
  __shared__ float s_w1[16][NGRP];
  __shared__ float s_att[16][NGRP];

  const int n   = blockIdx.x;
  const int tid = threadIdx.x;
  const int r   = tid >> 4, seg = tid & 15;

  if (tid < 16) {
    int id = refidx[n * NSMP + tid];
    float msk = (id >= 0) ? 1.f : 0.f;
    int safe = (id < 0) ? 0 : id;
    s_idx[tid]  = safe;
    s_mask[tid] = msk;
    s_pos[tid][0] = (xyz[safe * 3 + 0] - xyz[n * 3 + 0]) * msk;
    s_pos[tid][1] = (xyz[safe * 3 + 1] - xyz[n * 3 + 1]) * msk;
    s_pos[tid][2] = (xyz[safe * 3 + 2] - xyz[n * 3 + 2]) * msk;
  }
  __syncthreads();

  // pe-MLP layer 1 (pre-LN): h[s][c] = pos[s] . Wp1[:,c] + bp1[c]
  float px = s_pos[r][0], py = s_pos[r][1], pz = s_pos[r][2];
  float hreg[16];
  float sum = 0.f, sq = 0.f;
  #pragma unroll
  for (int i = 0; i < 16; ++i) {
    int c = seg * 16 + i;
    float h = px * Wp1[c] + py * Wp1[CH + c] + pz * Wp1[2 * CH + c] + bp1[c];
    hreg[i] = h; sum += h; sq += h * h;
  }
  s_red[0][r][seg] = sum;
  s_red[1][r][seg] = sq;
  __syncthreads();
  if (tid < 16) {
    float s1 = 0.f, s2 = 0.f;
    #pragma unroll
    for (int j = 0; j < 16; ++j) { s1 += s_red[0][tid][j]; s2 += s_red[1][tid][j]; }
    float mu = s1 * (1.f / CH);
    float var = s2 * (1.f / CH) - mu * mu;
    s_mu[tid] = mu;
    s_rs[tid] = rsqrtf(var + 1e-5f);
  }
  __syncthreads();
  {
    float mu = s_mu[r], rs = s_rs[r];
    #pragma unroll
    for (int i = 0; i < 16; ++i) {
      int c = seg * 16 + i;
      float y = (hreg[i] - mu) * rs * gp[c] + betap[c];
      s_hid[r][c] = (__bf16)fmaxf(y, 0.f);
    }
  }
  __syncthreads();

  // peb = hid @ Wp2 + bp2 via WMMA (16x256 @ 256x256)
  const int lane = tid & 31, wave = tid >> 5;
  const int m    = lane & 15;
  const int koff = (lane & 16) ? 8 : 0;
  const int nc0  = wave * 32 + m;
  const int nc1  = nc0 + 16;
  v8f acc0 = {}, acc1 = {};
  #pragma unroll
  for (int ks = 0; ks < 8; ++ks) {
    int kb = ks * 32 + koff;
    AFrag a, b0, b1;
    a.h[0]  = *(const v8bf*)&s_hid[m][kb];
    a.h[1]  = *(const v8bf*)&s_hid[m][kb + 16];
    b0.h[0] = *(const v8bf*)&Wp2bf[(size_t)nc0 * CH + kb];
    b0.h[1] = *(const v8bf*)&Wp2bf[(size_t)nc0 * CH + kb + 16];
    b1.h[0] = *(const v8bf*)&Wp2bf[(size_t)nc1 * CH + kb];
    b1.h[1] = *(const v8bf*)&Wp2bf[(size_t)nc1 * CH + kb + 16];
    acc0 = wmma_bf16(a, b0, acc0);
    acc1 = wmma_bf16(a, b1, acc1);
  }
  const int orow = (lane & 16) ? 8 : 0;
  #pragma unroll
  for (int rr = 0; rr < 8; ++rr) {
    s_peb[orow + rr][nc0] = acc0[rr] + bp2[nc0];
    s_peb[orow + rr][nc1] = acc1[rr] + bp2[nc1];
  }
  __syncthreads();

  // gather: val_g = val[idx]*mask + peb ; relation = keyf[idx]*mask - query[n] + peb
  {
    int   gid = s_idx[r];
    float msk = s_mask[r];
    const float* vrow = val   + (size_t)gid * CH;
    const float* krow = keyf  + (size_t)gid * CH;
    const float* qrow = query + (size_t)n   * CH;
    #pragma unroll
    for (int i = 0; i < 16; ++i) {
      int c = seg + 16 * i;
      float pe = s_peb[r][c];
      s_valg[r][c] = vrow[c] * msk + pe;
      s_peb[r][c]  = krow[c] * msk - qrow[c] + pe;  // relation_qk
    }
  }
  __syncthreads();

  // weight encoding layer 1: relation @ Ww1 + bw1  (16 x 8 outputs)
  if (tid < 16 * NGRP) {
    int s = tid >> 3, g = tid & 7;
    float acc = bw1[g];
    for (int c = 0; c < CH; ++c) acc += s_peb[s][c] * Ww1[c * NGRP + g];
    s_w1[s][g] = acc;
  }
  __syncthreads();

  // LN over G + ReLU + @Ww2 + bw2  (per neighbor row)
  if (tid < 16) {
    int s = tid;
    float mu = 0.f;
    #pragma unroll
    for (int g = 0; g < NGRP; ++g) mu += s_w1[s][g];
    mu *= (1.f / NGRP);
    float var = 0.f;
    #pragma unroll
    for (int g = 0; g < NGRP; ++g) { float d = s_w1[s][g] - mu; var += d * d; }
    var *= (1.f / NGRP);
    float rs = rsqrtf(var + 1e-5f);
    float h[NGRP];
    #pragma unroll
    for (int g = 0; g < NGRP; ++g)
      h[g] = fmaxf((s_w1[s][g] - mu) * rs * gw[g] + betaw[g], 0.f);
    #pragma unroll
    for (int g = 0; g < NGRP; ++g) {
      float acc = bw2[g];
      #pragma unroll
      for (int g2 = 0; g2 < NGRP; ++g2) acc += h[g2] * Ww2[g2 * NGRP + g];
      s_att[s][g] = acc;
    }
  }
  __syncthreads();

  // softmax over the 16 neighbors, per group, then mask
  if (tid < NGRP) {
    int g = tid;
    float mx = -1e30f;
    #pragma unroll
    for (int s = 0; s < NSMP; ++s) mx = fmaxf(mx, s_att[s][g]);
    float e[NSMP];
    float den = 0.f;
    #pragma unroll
    for (int s = 0; s < NSMP; ++s) { e[s] = __expf(s_att[s][g] - mx); den += e[s]; }
    float inv = 1.f / den;
    #pragma unroll
    for (int s = 0; s < NSMP; ++s) s_att[s][g] = e[s] * inv * s_mask[s];
  }
  __syncthreads();

  // grouped einsum: out[n][c] = sum_s att[s][c/32] * valg[s][c]
  {
    int c = tid;
    int g = c >> 5;   // C/G = 32
    float acc = 0.f;
    #pragma unroll
    for (int s = 0; s < NSMP; ++s) acc += s_att[s][g] * s_valg[s][c];
    out[(size_t)n * CH + c] = acc;
  }
}

// ---------------------------------------------------------------------------
extern "C" void kernel_launch(void* const* d_in, const int* in_sizes, int n_in,
                              void* d_out, int out_size, void* d_ws, size_t ws_size,
                              hipStream_t stream) {
  const float* q     = (const float*)d_in[0];
  const float* k     = (const float*)d_in[1];
  const float* v     = (const float*)d_in[2];
  const float* xyz   = (const float*)d_in[3];
  const int*   ridx  = (const int*)d_in[4];
  const float* Wq    = (const float*)d_in[5];
  const float* bq    = (const float*)d_in[6];
  const float* gq    = (const float*)d_in[7];
  const float* betaq = (const float*)d_in[8];
  const float* Wk    = (const float*)d_in[9];
  const float* bk    = (const float*)d_in[10];
  const float* gk    = (const float*)d_in[11];
  const float* betak = (const float*)d_in[12];
  const float* Wv    = (const float*)d_in[13];
  const float* bv    = (const float*)d_in[14];
  const float* Wp1   = (const float*)d_in[15];
  const float* bp1   = (const float*)d_in[16];
  const float* gp    = (const float*)d_in[17];
  const float* betap = (const float*)d_in[18];
  const float* Wp2   = (const float*)d_in[19];
  const float* bp2   = (const float*)d_in[20];
  const float* Ww1   = (const float*)d_in[21];
  const float* bw1   = (const float*)d_in[22];
  const float* gw    = (const float*)d_in[23];
  const float* betaw = (const float*)d_in[24];
  const float* Ww2   = (const float*)d_in[25];
  const float* bw2   = (const float*)d_in[26];
  float* out = (float*)d_out;

  // workspace layout: query | keyf | val (f32, 32 MB each) | 4 bf16 weights
  float* ws      = (float*)d_ws;
  float* f_query = ws;
  float* f_key   = ws + (size_t)NPTS * CH;
  float* f_val   = ws + 2 * (size_t)NPTS * CH;
  __bf16* wbf    = (__bf16*)(ws + 3 * (size_t)NPTS * CH);
  __bf16* Wp2bf  = wbf + 3 * (size_t)CH * CH;

  cvt_w_kernel<<<dim3(CH, 4), CH, 0, stream>>>(Wq, Wk, Wv, Wp2, wbf);

  proj_kernel<<<dim3(NPTS / 16, 3), 256, 0, stream>>>(
      q, k, v, wbf, bq, bk, bv, gq, betaq, gk, betak, f_query, f_key, f_val);

  attn_kernel<<<dim3(NPTS), 256, 0, stream>>>(
      f_query, f_key, f_val, xyz, ridx,
      Wp1, bp1, gp, betap, Wp2bf, bp2,
      Ww1, bw1, gw, betaw, Ww2, bw2, out);
}
